// AttentionBranch_4767413698993
// MI455X (gfx1250) — matrix operand order repaired and checked by reading, compiled, not run on hardware
//
#include <hip/hip_runtime.h>
#include <hip/hip_bf16.h>

// ---------------------------------------------------------------------------
// Swin-style window-attention transformer, MI455X (gfx1250, wave32, WMMA).
// Activations: fp32 residual stream; all GEMM operands bf16; fp32 accum via
// v_wmma_f32_16x16x32_bf16. Shift handled by index math in attention only.
// GEMM: 128x128 block tile, 32x64 wave tile (8 WMMA / K-step), double-buffered
// LDS staging -> one barrier per K-step.
// ---------------------------------------------------------------------------

#define TOKS   262144   // 4 * 256 * 256
#define CDIM   256
#define NHEAD  8
#define HDIM   32
#define HWN    65536
#define NLAY   6

typedef __attribute__((ext_vector_type(16))) __bf16 v16bf;
typedef __attribute__((ext_vector_type(8)))  float  v8f;

union FragU { v16bf v; unsigned int u32[8]; unsigned short u16[16]; };

__device__ __forceinline__ unsigned short f2bf(float f) {
  union { float f; unsigned int u; } c; c.f = f;
  unsigned int r = c.u + 0x7FFFu + ((c.u >> 16) & 1u);   // round-to-nearest-even
  return (unsigned short)(r >> 16);
}

__device__ __forceinline__ v8f wmma_bf16(v16bf a, v16bf b, v8f c) {
  // (neg_a, A, neg_b, B, c_mod, C, reuse_a, reuse_b)
  return __builtin_amdgcn_wmma_f32_16x16x32_bf16(false, a, false, b, (short)0, c,
                                                 false, false);
}

// A fragment 16x32 bf16 from row-major LDS [m][k], stride ld (elements).
// Contiguous k order used here (frag_b_* match): lanes 0-15 M=lane (K 0..15), lanes 16-31 same M (K 16..31).
__device__ __forceinline__ v16bf frag_a(const unsigned short* s, int mbase, int ld) {
  const int l  = threadIdx.x & 31;
  const int m  = mbase + (l & 15);
  const int h8 = (l >> 4) * 16;
  FragU f;
#pragma unroll
  for (int r = 0; r < 8; ++r) {
    const int k0 = 2 * r + h8;  // even -> dword load
    f.u32[r] = *(const unsigned int*)(s + m * ld + k0);
  }
  return f.v;
}

// B fragment 32x16 from k-major LDS [k][n] (stride ld): K pairs not contiguous.
__device__ __forceinline__ v16bf frag_b_kmaj(const unsigned short* s, int nbase, int ld) {
  const int l  = threadIdx.x & 31;
  const int n  = nbase + (l & 15);
  const int kh = (l >> 4) * 16;
  FragU f;
#pragma unroll
  for (int r = 0; r < 8; ++r) {
    f.u16[2 * r]     = s[(kh + 2 * r)     * ld + n];
    f.u16[2 * r + 1] = s[(kh + 2 * r + 1) * ld + n];
  }
  return f.v;
}

// B fragment 32x16 from n-major LDS [n][k] (stride ld): K pairs contiguous.
__device__ __forceinline__ v16bf frag_b_nmaj(const unsigned short* s, int nbase,
                                             int kbase, int ld) {
  const int l  = threadIdx.x & 31;
  const int n  = nbase + (l & 15);
  const int kh = kbase + (l >> 4) * 16;
  FragU f;
#pragma unroll
  for (int r = 0; r < 8; ++r)
    f.u32[r] = *(const unsigned int*)(s + n * ld + kh + 2 * r);
  return f.v;
}

// ---------------------------------------------------------------------------
// Fused GEMM: out = act(A[M,K](bf16) @ B[K,N](bf16) + bias (+ residual fp32))
// 256 thr = 8 waves; block 128x128; wave 32x64 (2x4 WMMA accums); BK=32;
// double-buffered LDS, one barrier per K-step; prefetch of K+64 tile.
// ---------------------------------------------------------------------------
template <bool GELU, bool HAS_RES, bool OUT_BF16, bool OUT_F32>
__global__ __launch_bounds__(256, 2) void k_gemm(
    const unsigned short* __restrict__ A, const unsigned short* __restrict__ Bw,
    const float* __restrict__ bias, const float* __restrict__ res,
    unsigned short* __restrict__ ob, float* __restrict__ of,
    int M, int N, int K)
{
  __shared__ unsigned short sA[2 * 128 * 40];   // 2 x (128x32, stride 40)
  __shared__ unsigned short sB[2 * 32 * 136];   // 2 x (32x128, stride 136)
  const int tid  = threadIdx.x;
  const int m0   = blockIdx.y * 128;
  const int n0   = blockIdx.x * 128;
  const int wave = tid >> 5;
  const int mb   = (wave & 3) * 32;
  const int nb   = (wave >> 2) * 64;

  v8f acc[2][4] = {};

  auto stage = [&](int buf, int k0) {
    unsigned short* dA = sA + buf * 5120;
    unsigned short* dB = sB + buf * 4352;
    // A tile: 128 rows x 32 bf16 = 512 x 16B chunks, coalesced.
    for (int idx = tid; idx < 512; idx += 256) {
      const int r = idx >> 2, q = idx & 3;
      *(uint4*)(dA + r * 40 + q * 8) =
          *(const uint4*)(A + (long)(m0 + r) * K + k0 + q * 8);
    }
    // B tile: 32 rows x 128 bf16 = 512 x 16B chunks.
    for (int idx = tid; idx < 512; idx += 256) {
      const int r = idx >> 4, q = idx & 15;
      *(uint4*)(dB + r * 136 + q * 8) =
          *(const uint4*)(Bw + (long)(k0 + r) * N + n0 + q * 8);
    }
  };

  int buf = 0;
  stage(0, 0);
  __syncthreads();

  for (int k0 = 0; k0 < K; k0 += 32) {
    if (k0 + 32 < K) {
      stage(buf ^ 1, k0 + 32);                       // fill other buffer
      if (k0 + 64 < K) {                              // prefetch K+64 tile
        __builtin_prefetch(A + (long)(m0 + (tid >> 1)) * K + k0 + 64, 0, 0);
        __builtin_prefetch(Bw + (long)(k0 + 64 + (tid >> 3)) * N + n0, 0, 0);
      }
    }
    const unsigned short* cA = sA + buf * 5120;
    const unsigned short* cB = sB + buf * 4352;
    const v16bf a0 = frag_a(cA, mb,      40);
    const v16bf a1 = frag_a(cA, mb + 16, 40);
    v16bf bfr[4];
#pragma unroll
    for (int j = 0; j < 4; ++j) bfr[j] = frag_b_kmaj(cB, nb + j * 16, 136);
#pragma unroll
    for (int j = 0; j < 4; ++j) {
      acc[0][j] = wmma_bf16(a0, bfr[j], acc[0][j]);
      acc[1][j] = wmma_bf16(a1, bfr[j], acc[1][j]);
    }
    __syncthreads();
    buf ^= 1;
  }

  // Epilogue. D layout: lane=col (n = l&15), row = r + 8*(l>>4).
  const int l  = tid & 31;
  const int hf = l >> 4;
#pragma unroll
  for (int j = 0; j < 4; ++j) {
    const int col = n0 + nb + j * 16 + (l & 15);
    const float bj = bias[col];
#pragma unroll
    for (int i = 0; i < 2; ++i) {
#pragma unroll
      for (int r = 0; r < 8; ++r) {
        const int row = m0 + mb + i * 16 + r + 8 * hf;
        float v = acc[i][j][r] + bj;
        if constexpr (HAS_RES) v += res[(long)row * N + col];
        if constexpr (GELU)    v = 0.5f * v * (1.f + erff(v * 0.70710678118f));
        if constexpr (OUT_F32)  of[(long)row * N + col] = v;
        if constexpr (OUT_BF16) ob[(long)row * N + col] = f2bf(v);
      }
    }
  }
}

// ---------------------------------------------------------------------------
// Window attention: one block (128 thr = 4 waves) per (window, head).
// Shift folded into token gather/scatter. qkv layout: [tok][3][NH][HD] bf16.
// ---------------------------------------------------------------------------
__global__ __launch_bounds__(128, 4) void k_attn(
    const unsigned short* __restrict__ qkv, unsigned short* __restrict__ out,
    int shift)
{
  __shared__ unsigned short sQ[64 * 32];
  __shared__ unsigned short sK[64 * 32];
  __shared__ unsigned short sV[64 * 32];
  __shared__ unsigned short sP[64 * 64];
  __shared__ float sInv[64];

  const int tid  = threadIdx.x;
  const int win  = blockIdx.x;            // 0..1023
  const int head = blockIdx.y;            // 0..7
  const int b    = blockIdx.z;            // 0..3
  const int wy   = win >> 5, wx = win & 31;

  // Stage Q,K,V (3 x 64 rows x 4 x 16B chunks) with shifted-window gather.
  for (int idx = tid; idx < 768; idx += 128) {
    const int mat = idx >> 8;             // 0=Q 1=K 2=V
    const int rem = idx & 255;
    const int row = rem >> 2, q = rem & 3;
    const int gr  = (wy * 8 + (row >> 3) + shift) & 255;
    const int gc  = (wx * 8 + (row & 7)  + shift) & 255;
    const long tok = ((long)b << 16) | (gr << 8) | gc;
    unsigned short* dst = (mat == 0 ? sQ : (mat == 1 ? sK : sV)) + row * 32 + q * 8;
    *(uint4*)dst =
        *(const uint4*)(qkv + tok * 768 + mat * 256 + head * 32 + q * 8);
  }
  __syncthreads();

  const int wave  = tid >> 5;
  const int mbase = wave * 16;            // 16 query rows per wave
  const int l     = tid & 31;
  const int hf    = l >> 4;

  // scores[16x64] = Q @ K^T : B matrix = K^T, n-major LDS gives dword pairs.
  const v16bf aq = frag_a(sQ, mbase, 32);
  v8f sc[4];
#pragma unroll
  for (int nt = 0; nt < 4; ++nt) {
    v8f z = {};
    sc[nt] = wmma_bf16(aq, frag_b_nmaj(sK, nt * 16, 0, 32), z);
  }

  // Softmax over 64 keys. Row r+8*hf lives in one 16-lane half -> xor 1/2/4/8.
  const float scale = 0.17677669529663689f;   // 1/sqrt(32)
  float rsum[8];
#pragma unroll
  for (int r = 0; r < 8; ++r) {
    float m = sc[0][r] * scale;
#pragma unroll
    for (int nt = 1; nt < 4; ++nt) m = fmaxf(m, sc[nt][r] * scale);
    m = fmaxf(m, __shfl_xor(m, 1, 32));
    m = fmaxf(m, __shfl_xor(m, 2, 32));
    m = fmaxf(m, __shfl_xor(m, 4, 32));
    m = fmaxf(m, __shfl_xor(m, 8, 32));
    float s = 0.f;
#pragma unroll
    for (int nt = 0; nt < 4; ++nt) {
      const float p = __expf(sc[nt][r] * scale - m);
      sc[nt][r] = p;
      s += p;
    }
    s += __shfl_xor(s, 1, 32);
    s += __shfl_xor(s, 2, 32);
    s += __shfl_xor(s, 4, 32);
    s += __shfl_xor(s, 8, 32);
    rsum[r] = s;
  }

  // D-layout -> A-layout via LDS (own 16 rows only; intra-wave, no barrier).
#pragma unroll
  for (int r = 0; r < 8; ++r) {
    const int row = mbase + r + 8 * hf;
#pragma unroll
    for (int nt = 0; nt < 4; ++nt)
      sP[row * 64 + nt * 16 + (l & 15)] = f2bf(sc[nt][r]);
    if ((l & 15) == 0) sInv[row] = 1.f / rsum[r];
  }

  // out[16x32] = P @ V, K-dim 64 in two WMMA steps.
  v8f o[2] = {};
#pragma unroll
  for (int kt = 0; kt < 2; ++kt) {
    const v16bf ap = frag_a(sP + kt * 32, mbase, 64);
#pragma unroll
    for (int nt = 0; nt < 2; ++nt)
      o[nt] = wmma_bf16(ap, frag_b_kmaj(sV + kt * 32 * 32, nt * 16, 32), o[nt]);
  }

  // Scale by 1/rowsum, scatter back with shifted indices (bf16 for proj GEMM).
#pragma unroll
  for (int r = 0; r < 8; ++r) {
    const int row = mbase + r + 8 * hf;
    const float inv = sInv[row];
    const int gr = (wy * 8 + (row >> 3) + shift) & 255;
    const int gc = (wx * 8 + (row & 7)  + shift) & 255;
    const long tok = ((long)b << 16) | (gr << 8) | gc;
#pragma unroll
    for (int nt = 0; nt < 2; ++nt)
      out[tok * 256 + head * 32 + nt * 16 + (l & 15)] = f2bf(o[nt][r] * inv);
  }
}

// ---------------------------------------------------------------------------
// LayerNorm fp32 -> bf16, wave-per-row (32 lanes x 8 elems), C=256.
// ---------------------------------------------------------------------------
__global__ __launch_bounds__(256) void k_ln(const float* __restrict__ x,
    const float* __restrict__ g, const float* __restrict__ bta,
    unsigned short* __restrict__ out)
{
  const int row = blockIdx.x * 8 + (threadIdx.x >> 5);
  const int l   = threadIdx.x & 31;
  const float4* p = (const float4*)(x + (long)row * 256 + l * 8);
  const float4 u = p[0], w = p[1];
  const float v[8] = {u.x, u.y, u.z, u.w, w.x, w.y, w.z, w.w};
  float s = 0.f, q = 0.f;
#pragma unroll
  for (int e = 0; e < 8; ++e) { s += v[e]; q += v[e] * v[e]; }
  for (int m = 1; m < 32; m <<= 1) {
    s += __shfl_xor(s, m, 32);
    q += __shfl_xor(q, m, 32);
  }
  const float mean = s * (1.f / 256.f);
  const float rs   = rsqrtf(q * (1.f / 256.f) - mean * mean + 1e-5f);
  unsigned int pk[4];
#pragma unroll
  for (int e = 0; e < 8; e += 2) {
    const float a0 = (v[e]     - mean) * rs * g[l * 8 + e]     + bta[l * 8 + e];
    const float a1 = (v[e + 1] - mean) * rs * g[l * 8 + e + 1] + bta[l * 8 + e + 1];
    pk[e >> 1] = (unsigned)f2bf(a0) | ((unsigned)f2bf(a1) << 16);
  }
  *(uint4*)(out + (long)row * 256 + l * 8) = make_uint4(pk[0], pk[1], pk[2], pk[3]);
}

// fp32 -> bf16 copy (weights + final activation cast)
__global__ void k_cvt(const float* __restrict__ in, unsigned short* __restrict__ out,
                      long n) {
  const long i = (long)blockIdx.x * blockDim.x + threadIdx.x;
  if (i < n) out[i] = f2bf(in[i]);
}

// output_proj_w[o,i] -> bf16 B[k=i, n=o] (transpose during one-time convert)
__global__ void k_cvtT256(const float* __restrict__ in,
                          unsigned short* __restrict__ out) {
  const int k = blockIdx.x, n = threadIdx.x;
  out[k * 256 + n] = f2bf(in[n * 256 + k]);
}

// 1x1 conv input projection + positional embedding -> token-major fp32
__global__ __launch_bounds__(256) void k_inproj(const float* __restrict__ x,
    const float* __restrict__ W, const float* __restrict__ bias,
    const float* __restrict__ pos, float* __restrict__ X)
{
  const long t = blockIdx.x;
  const int b = (int)(t >> 16), hw = (int)(t & 65535);
  const int c = threadIdx.x;
  const float* xb = x + (long)b * 3 * 65536 + hw;
  const float o = xb[0] * W[c * 3] + xb[65536] * W[c * 3 + 1] +
                  xb[131072] * W[c * 3 + 2] + bias[c] +
                  pos[(long)c * 65536 + hw];
  X[t * 256 + c] = o;
}

// [b][hw][c] fp32 -> [b][c][hw] (final NCHW output), tiled 32x32 transpose
__global__ __launch_bounds__(256) void k_to_nchw(const float* __restrict__ in,
                                                 float* __restrict__ out) {
  __shared__ float t[32][33];
  const int b = blockIdx.z;
  const int hw0 = blockIdx.x * 32, c0 = blockIdx.y * 32;
  const int tx = threadIdx.x, ty = threadIdx.y;    // (32,8)
#pragma unroll
  for (int i = 0; i < 4; ++i)
    t[ty + 8 * i][tx] = in[((long)b * 65536 + hw0 + ty + 8 * i) * 256 + c0 + tx];
  __syncthreads();
#pragma unroll
  for (int i = 0; i < 4; ++i)
    out[((long)b * 256 + c0 + ty + 8 * i) * 65536 + hw0 + tx] = t[tx][ty + 8 * i];
}

// ---------------------------------------------------------------------------
extern "C" void kernel_launch(void* const* d_in, const int* in_sizes, int n_in,
                              void* d_out, int out_size, void* d_ws, size_t ws_size,
                              hipStream_t stream)
{
  (void)in_sizes; (void)n_in; (void)out_size; (void)ws_size;

  const float* x_in = (const float*)d_in[0];
  const float* ipw  = (const float*)d_in[1];
  const float* ipb  = (const float*)d_in[2];
  const float* pos  = (const float*)d_in[3];
  const float* opw  = (const float*)d_in[4 + NLAY * 12];
  const float* opb  = (const float*)d_in[5 + NLAY * 12];

  char* base = (char*)d_ws;
  size_t off = 0;
  auto wsalloc = [&](size_t bytes) {
    char* p = base + off;
    off = (off + bytes + 255) & ~(size_t)255;
    return p;
  };
  float*          wX   = (float*)wsalloc((size_t)TOKS * 256 * 4);   // residual A
  float*          wY   = (float*)wsalloc((size_t)TOKS * 256 * 4);   // residual B
  unsigned short* wN   = (unsigned short*)wsalloc((size_t)TOKS * 256 * 2);  // LN/attn bf16
  unsigned short* wBig = (unsigned short*)wsalloc((size_t)TOKS * 1024 * 2); // qkv / fc1
  unsigned short* wW   = (unsigned short*)wsalloc(((size_t)NLAY * 786432 + 65536) * 2);

  // One-time fp32 -> bf16 weight conversion (layer stride 786432 elements).
  for (int i = 0; i < NLAY; ++i) {
    unsigned short* dst = wW + (size_t)i * 786432;
    k_cvt<<<768,  256, 0, stream>>>((const float*)d_in[4 + i*12 + 2],  dst,          196608);
    k_cvt<<<256,  256, 0, stream>>>((const float*)d_in[4 + i*12 + 4],  dst + 196608, 65536);
    k_cvt<<<1024, 256, 0, stream>>>((const float*)d_in[4 + i*12 + 8],  dst + 262144, 262144);
    k_cvt<<<1024, 256, 0, stream>>>((const float*)d_in[4 + i*12 + 10], dst + 524288, 262144);
  }
  unsigned short* wOpT = wW + (size_t)NLAY * 786432;
  k_cvtT256<<<256, 256, 0, stream>>>(opw, wOpT);

  k_inproj<<<TOKS, 256, 0, stream>>>(x_in, ipw, ipb, pos, wX);

  const dim3 gQKV(768 / 128,  TOKS / 128);
  const dim3 gC  (256 / 128,  TOKS / 128);
  const dim3 gFC1(1024 / 128, TOKS / 128);

  for (int i = 0; i < NLAY; ++i) {
    const float* n1g  = (const float*)d_in[4 + i*12 + 0];
    const float* n1b  = (const float*)d_in[4 + i*12 + 1];
    const float* qkvb = (const float*)d_in[4 + i*12 + 3];
    const float* pjb  = (const float*)d_in[4 + i*12 + 5];
    const float* n2g  = (const float*)d_in[4 + i*12 + 6];
    const float* n2b  = (const float*)d_in[4 + i*12 + 7];
    const float* f1b  = (const float*)d_in[4 + i*12 + 9];
    const float* f2b  = (const float*)d_in[4 + i*12 + 11];
    unsigned short* Wl = wW + (size_t)i * 786432;
    const int shift = (i & 1) ? 4 : 0;

    k_ln<<<TOKS / 8, 256, 0, stream>>>(wX, n1g, n1b, wN);
    k_gemm<false, false, true, false><<<gQKV, 256, 0, stream>>>(
        wN, Wl, qkvb, nullptr, wBig, nullptr, TOKS, 768, 256);
    k_attn<<<dim3(1024, NHEAD, 4), 128, 0, stream>>>(wBig, wN, shift);
    k_gemm<false, true, false, true><<<gC, 256, 0, stream>>>(
        wN, Wl + 196608, pjb, wX, nullptr, wY, TOKS, 256, 256);
    k_ln<<<TOKS / 8, 256, 0, stream>>>(wY, n2g, n2b, wN);
    k_gemm<true, false, true, false><<<gFC1, 256, 0, stream>>>(
        wN, Wl + 262144, f1b, nullptr, wBig, nullptr, TOKS, 1024, 256);
    k_gemm<false, true, false, true><<<gC, 256, 0, stream>>>(
        wBig, Wl + 524288, f2b, wY, nullptr, wX, TOKS, 256, 1024);
  }

  // Output projection (WMMA) + permute to NCHW.
  k_cvt<<<TOKS, 256, 0, stream>>>(wX, wN, (long)TOKS * 256);
  k_gemm<false, false, false, true><<<gC, 256, 0, stream>>>(
      wN, wOpT, opb, nullptr, nullptr, wY, TOKS, 256, 256);
  k_to_nchw<<<dim3(2048, 8, 4), dim3(32, 8), 0, stream>>>(wY, (float*)d_out);
}
